// mqa_20598663152347
// MI455X (gfx1250) — matrix operand order repaired and checked by reading, not compiled, not run on hardware
//
#include <hip/hip_runtime.h>

#define BATCH 2
#define SEQ   2048
#define DIM   2048
#define NH    16
#define HD    128

typedef __attribute__((ext_vector_type(16))) __bf16 v16bf;
typedef __attribute__((ext_vector_type(2)))  __bf16 v2bf;
typedef __attribute__((ext_vector_type(8)))  float  v8f;
typedef __attribute__((ext_vector_type(8)))  unsigned v8u;

// Pack two f32 into two bf16 (one VALU op: v_cvt_pk_bf16_f32).
static __device__ __forceinline__ unsigned cvt_pk_bf16(float a, float b) {
#if __has_builtin(__builtin_amdgcn_cvt_pk_bf16_f32)
  v2bf p = __builtin_amdgcn_cvt_pk_bf16_f32(a, b);
  return __builtin_bit_cast(unsigned, p);
#else
  unsigned d;
  asm("v_cvt_pk_bf16_f32 %0, %1, %2" : "=v"(d) : "v"(a), "v"(b));
  return d;
#endif
}

static __device__ __forceinline__ uint4 pack8(const float4& lo, const float4& hi) {
  uint4 u;
  u.x = cvt_pk_bf16(lo.x, lo.y);
  u.y = cvt_pk_bf16(lo.z, lo.w);
  u.z = cvt_pk_bf16(hi.x, hi.y);
  u.w = cvt_pk_bf16(hi.z, hi.w);
  return u;
}

// ---------------------------------------------------------------------------
// GEMM core (shared shape): C = A[M,K] @ W[N,K]^T, 128x128 block tile,
// 256 threads / 8 waves, wave = 64x32 (8 WMMA tiles), double-buffered LDS,
// global_prefetch two K-steps ahead. Epilogue differs per kernel below.
// ---------------------------------------------------------------------------
#define GEMM_CORE(ACC)                                                        \
  __shared__ __align__(16) __bf16 As[2][128][40];                             \
  __shared__ __align__(16) __bf16 Ws[2][128][40];                             \
  const int tid  = threadIdx.x;                                               \
  const int lane = tid & 31;                                                  \
  const int wid  = tid >> 5;                                                  \
  const int g    = lane >> 4;                                                 \
  const int ln   = lane & 15;                                                 \
  const int wm   = wid >> 2;                                                  \
  const int wn   = wid & 3;                                                   \
  const int bm   = blockIdx.x * 128;                                          \
  const int bn   = blockIdx.y * 128;                                          \
  v8f ACC[4][2];                                                              \
  _Pragma("unroll") for (int mt = 0; mt < 4; ++mt)                            \
  _Pragma("unroll") for (int nt = 0; nt < 2; ++nt)                            \
  _Pragma("unroll") for (int i = 0; i < 8; ++i) ACC[mt][nt][i] = 0.0f;        \
  const int r  = tid >> 1;                                                    \
  const int c0 = (tid & 1) * 16;                                              \
  const float* arow = A + (size_t)(bm + r) * Kdim + c0;                       \
  const float* wrow = W + (size_t)(bn + r) * Kdim + c0;                       \
  auto stage = [&](int buf, int k0) {                                         \
    const float4* ag = (const float4*)(arow + k0);                            \
    const float4* wg = (const float4*)(wrow + k0);                            \
    float4 a0 = ag[0], a1 = ag[1], a2 = ag[2], a3 = ag[3];                    \
    float4 w0 = wg[0], w1 = wg[1], w2 = wg[2], w3 = wg[3];                    \
    *(uint4*)&As[buf][r][c0]     = pack8(a0, a1);                             \
    *(uint4*)&As[buf][r][c0 + 8] = pack8(a2, a3);                             \
    *(uint4*)&Ws[buf][r][c0]     = pack8(w0, w1);                             \
    *(uint4*)&Ws[buf][r][c0 + 8] = pack8(w2, w3);                             \
  };                                                                          \
  const int nk = Kdim >> 5;                                                   \
  stage(0, 0);                                                                \
  int buf = 0;                                                                \
  for (int kt = 0; kt < nk; ++kt) {                                           \
    __syncthreads();                                                          \
    if (kt + 2 < nk) {                                                        \
      __builtin_prefetch(arow + ((kt + 2) << 5), 0, 1);                       \
      __builtin_prefetch(wrow + ((kt + 2) << 5), 0, 1);                       \
    }                                                                         \
    if (kt + 1 < nk) stage(buf ^ 1, (kt + 1) << 5);                           \
    v16bf bfrag[2], afrag[4];                                                 \
    _Pragma("unroll") for (int nt = 0; nt < 2; ++nt) {                        \
      const __bf16* wr = &Ws[buf][wn * 32 + nt * 16 + ln][g * 16];            \
      _Pragma("unroll") for (int e = 0; e < 16; ++e) bfrag[nt][e] = wr[e];    \
    }                                                                         \
    _Pragma("unroll") for (int mt = 0; mt < 4; ++mt) {                        \
      const __bf16* ar = &As[buf][wm * 64 + mt * 16 + ln][0];                 \
      _Pragma("unroll") for (int e = 0; e < 8; ++e)                           \
        afrag[mt][e]     = ar[g * 16 + e];                                    \
      _Pragma("unroll") for (int e = 0; e < 8; ++e)                           \
        afrag[mt][8 + e] = ar[g * 16 + 8 + e];                                \
    }                                                                         \
    _Pragma("unroll") for (int mt = 0; mt < 4; ++mt)                          \
    _Pragma("unroll") for (int nt = 0; nt < 2; ++nt)                          \
      ACC[mt][nt] = __builtin_amdgcn_wmma_f32_16x16x32_bf16(                  \
          false, afrag[mt], false, bfrag[nt], (short)0, ACC[mt][nt],          \
          false, false);                                                      \
    buf ^= 1;                                                                 \
  }

// f32 output epilogue (Q projection, final O projection).
__global__ __launch_bounds__(256) void gemm_bias_f32out(
    const float* __restrict__ A, const float* __restrict__ W,
    const float* __restrict__ bias, float* __restrict__ C,
    int M, int N, int Kdim)
{
  GEMM_CORE(acc)
#pragma unroll
  for (int mt = 0; mt < 4; ++mt)
#pragma unroll
    for (int nt = 0; nt < 2; ++nt) {
      const int col = bn + wn * 32 + nt * 16 + ln;
      const float bv = bias[col];
#pragma unroll
      for (int i = 0; i < 8; ++i) {
        const int row = bm + wm * 64 + mt * 16 + i + 8 * g;
        C[(size_t)row * N + col] = acc[mt][nt][i] + bv;
      }
    }
}

// bf16 output epilogue (K/V projections). transposeOut==0: bf16 [M][N]
// row-major (K cache). transposeOut==1: bf16 [b][N][SEQ] (V pre-transposed
// for the attention PV operand; accumulator rows i,i+1 are adjacent tokens
// in the transposed layout -> one cvt_pk + one b32 store).
__global__ __launch_bounds__(256) void gemm_bias_bf16out(
    const float* __restrict__ A, const float* __restrict__ W,
    const float* __restrict__ bias, __bf16* __restrict__ C,
    int M, int N, int Kdim, int transposeOut)
{
  GEMM_CORE(acc)
#pragma unroll
  for (int mt = 0; mt < 4; ++mt)
#pragma unroll
    for (int nt = 0; nt < 2; ++nt) {
      const int col = bn + wn * 32 + nt * 16 + ln;
      const float bv = bias[col];
      if (transposeOut == 0) {
#pragma unroll
        for (int i = 0; i < 8; i += 2) {
          const int row = bm + wm * 64 + mt * 16 + i + 8 * g;
          v2bf pp = __builtin_bit_cast(
              v2bf, cvt_pk_bf16(acc[mt][nt][i] + bv, acc[mt][nt][i + 1] + bv));
          C[(size_t)row * N + col]       = pp.x;
          C[(size_t)(row + 1) * N + col] = pp.y;
        }
      } else {
#pragma unroll
        for (int i = 0; i < 8; i += 2) {
          const int row = bm + wm * 64 + mt * 16 + i + 8 * g;  // token (even)
          const int bb  = row >> 11;          // row / SEQ
          const int rb  = row & (SEQ - 1);    // row % SEQ
          unsigned u = cvt_pk_bf16(acc[mt][nt][i] + bv, acc[mt][nt][i + 1] + bv);
          *(unsigned*)&C[((size_t)bb * N + col) * SEQ + rb] = u;
        }
      }
    }
}

// ---------------------------------------------------------------------------
// Flash-style MQA attention. Grid: (SEQ/128, BATCH*NH), 256 threads (8 waves).
// K/V arrive as bf16 (V already [b][hd][s]); staging is pure b128 copies.
// ---------------------------------------------------------------------------
__global__ __launch_bounds__(256) void mqa_flash_attn(
    const float* __restrict__ Q, const __bf16* __restrict__ Kbf,
    const __bf16* __restrict__ Vtbf, float* __restrict__ O)
{
  __shared__ __align__(16) __bf16 Ks[64][HD + 16];   // 288-byte rows
  __shared__ __align__(16) __bf16 Vt[HD][64 + 8];    // [hd][key]
  __shared__ __align__(16) __bf16 Ps[8][16][64 + 8]; // per-wave P scratch

  const int tid  = threadIdx.x;
  const int lane = tid & 31;
  const int wid  = tid >> 5;
  const int g    = lane >> 4;
  const int ln   = lane & 15;
  const int b    = blockIdx.y >> 4;
  const int h    = blockIdx.y & 15;
  const int q0   = blockIdx.x * 128 + wid * 16;

  const float scale = 0.08838834764831845f;   // 1/sqrt(128)

  // Q fragments for this wave's 16 rows (scaled, bf16), 4 k-steps over HD.
  v16bf qa[4];
  {
    const float* qrow = Q + ((size_t)(b * SEQ + q0 + ln)) * DIM + h * HD;
#pragma unroll
    for (int ks = 0; ks < 4; ++ks) {
      const float4* p0 = (const float4*)(qrow + ks * 32 + g * 16);
      const float4* p1 = (const float4*)(qrow + ks * 32 + g * 16 + 8);
      float4 a0 = p0[0], a1 = p0[1], b0 = p1[0], b1 = p1[1];
      v8u u;
      u[0] = cvt_pk_bf16(a0.x * scale, a0.y * scale);
      u[1] = cvt_pk_bf16(a0.z * scale, a0.w * scale);
      u[2] = cvt_pk_bf16(a1.x * scale, a1.y * scale);
      u[3] = cvt_pk_bf16(a1.z * scale, a1.w * scale);
      u[4] = cvt_pk_bf16(b0.x * scale, b0.y * scale);
      u[5] = cvt_pk_bf16(b0.z * scale, b0.w * scale);
      u[6] = cvt_pk_bf16(b1.x * scale, b1.y * scale);
      u[7] = cvt_pk_bf16(b1.z * scale, b1.w * scale);
      qa[ks] = __builtin_bit_cast(v16bf, u);
    }
  }

  float m[8], l[8];
  v8f oacc[8];
#pragma unroll
  for (int i = 0; i < 8; ++i) { m[i] = -3.0e38f; l[i] = 0.0f; }
#pragma unroll
  for (int t = 0; t < 8; ++t)
#pragma unroll
    for (int i = 0; i < 8; ++i) oacc[t][i] = 0.0f;

  const __bf16* kbb = Kbf  + (size_t)b * SEQ * HD;   // [s][hd]
  const __bf16* vtb = Vtbf + (size_t)b * HD * SEQ;   // [hd][s]

  // Staging maps (fixed per thread): pure bf16 copies, 64B per thread each.
  const int krr = tid >> 2;            // K: key row 0..63
  const int kcc = (tid & 3) * 32;      // K: hd col start (32 bf16)
  const int vhd = tid >> 1;            // V: hd row 0..127
  const int vcc = (tid & 1) * 32;      // V: key col start (32 bf16)

  for (int kb = 0; kb < SEQ; kb += 64) {
    {
      const uint4* kg = (const uint4*)(kbb + (size_t)(kb + krr) * HD + kcc);
      uint4 k0 = kg[0], k1 = kg[1], k2 = kg[2], k3 = kg[3];
      *(uint4*)&Ks[krr][kcc]      = k0;
      *(uint4*)&Ks[krr][kcc + 8]  = k1;
      *(uint4*)&Ks[krr][kcc + 16] = k2;
      *(uint4*)&Ks[krr][kcc + 24] = k3;

      const uint4* vg = (const uint4*)(vtb + (size_t)vhd * SEQ + kb + vcc);
      uint4 v0 = vg[0], v1 = vg[1], v2 = vg[2], v3 = vg[3];
      *(uint4*)&Vt[vhd][vcc]      = v0;
      *(uint4*)&Vt[vhd][vcc + 8]  = v1;
      *(uint4*)&Vt[vhd][vcc + 16] = v2;
      *(uint4*)&Vt[vhd][vcc + 24] = v3;
    }
    __syncthreads();

    // S = Q @ Kb^T  (4 key tiles x 4 hd k-steps)
    v8f sacc[4];
#pragma unroll
    for (int nt = 0; nt < 4; ++nt)
#pragma unroll
      for (int i = 0; i < 8; ++i) sacc[nt][i] = 0.0f;
#pragma unroll
    for (int ks = 0; ks < 4; ++ks) {
#pragma unroll
      for (int nt = 0; nt < 4; ++nt) {
        v16bf kfrag;
        const __bf16* krow = &Ks[nt * 16 + ln][ks * 32 + g * 16];
#pragma unroll
        for (int e = 0; e < 16; ++e) kfrag[e] = krow[e];
        sacc[nt] = __builtin_amdgcn_wmma_f32_16x16x32_bf16(
            false, qa[ks], false, kfrag, (short)0, sacc[nt], false, false);
      }
    }

    // Online softmax per row (rows i+8g live on 16-lane half-groups).
    float p[4][8];
#pragma unroll
    for (int i = 0; i < 8; ++i) {
      float tmax = sacc[0][i];
#pragma unroll
      for (int nt = 1; nt < 4; ++nt) tmax = fmaxf(tmax, sacc[nt][i]);
#pragma unroll
      for (int mk = 8; mk >= 1; mk >>= 1)
        tmax = fmaxf(tmax, __shfl_xor(tmax, mk, 32));
      const float mnew  = fmaxf(m[i], tmax);
      const float alpha = __expf(m[i] - mnew);
      float rsum = 0.0f;
#pragma unroll
      for (int nt = 0; nt < 4; ++nt) {
        const float pv = __expf(sacc[nt][i] - mnew);
        p[nt][i] = pv;
        rsum += pv;
      }
#pragma unroll
      for (int mk = 8; mk >= 1; mk >>= 1)
        rsum += __shfl_xor(rsum, mk, 32);
      l[i] = l[i] * alpha + rsum;
      m[i] = mnew;
#pragma unroll
      for (int t = 0; t < 8; ++t) oacc[t][i] *= alpha;
    }

    // Re-layout P (C-layout) -> A-layout via per-wave LDS scratch.
    // No barrier needed: Ps is wave-private and DS ops are in-order per wave.
#pragma unroll
    for (int nt = 0; nt < 4; ++nt)
#pragma unroll
      for (int i = 0; i < 8; i += 2) {
        v2bf pp = __builtin_bit_cast(v2bf, cvt_pk_bf16(p[nt][i], p[nt][i + 1]));
        Ps[wid][i + 8 * g][nt * 16 + ln]     = pp.x;
        Ps[wid][i + 1 + 8 * g][nt * 16 + ln] = pp.y;
      }

    // O += P @ Vb  (2 key k-steps x 8 hd tiles); Vt rows are contiguous keys.
#pragma unroll
    for (int ks2 = 0; ks2 < 2; ++ks2) {
      v16bf pfrag;
      const __bf16* pr = &Ps[wid][ln][ks2 * 32];
#pragma unroll
      for (int e = 0; e < 8; ++e) pfrag[e]     = pr[g * 16 + e];
#pragma unroll
      for (int e = 0; e < 8; ++e) pfrag[8 + e] = pr[g * 16 + 8 + e];
#pragma unroll
      for (int t = 0; t < 8; ++t) {
        v16bf vfrag;
        const __bf16* vrow = &Vt[t * 16 + ln][ks2 * 32 + g * 16];
#pragma unroll
        for (int e = 0; e < 16; ++e) vfrag[e] = vrow[e];
        oacc[t] = __builtin_amdgcn_wmma_f32_16x16x32_bf16(
            false, pfrag, false, vfrag, (short)0, oacc[t], false, false);
      }
    }
    __syncthreads();
  }

  // Normalize and write O (interleaved back to [b, s, nh*hd]).
#pragma unroll
  for (int i = 0; i < 8; ++i) {
    const float inv = 1.0f / l[i];
    const int row = q0 + i + 8 * g;
    float* orow = O + ((size_t)(b * SEQ + row)) * DIM + h * HD;
#pragma unroll
    for (int t = 0; t < 8; ++t)
      orow[t * 16 + ln] = oacc[t][i] * inv;
  }
}

// ---------------------------------------------------------------------------
extern "C" void kernel_launch(void* const* d_in, const int* in_sizes, int n_in,
                              void* d_out, int out_size, void* d_ws, size_t ws_size,
                              hipStream_t stream) {
  const float* x  = (const float*)d_in[0];
  const float* wq = (const float*)d_in[1];
  const float* bq = (const float*)d_in[2];
  const float* wk = (const float*)d_in[3];
  const float* bk = (const float*)d_in[4];
  const float* wv = (const float*)d_in[5];
  const float* bv = (const float*)d_in[6];
  const float* wo = (const float*)d_in[7];
  const float* bo = (const float*)d_in[8];
  float* out = (float*)d_out;

  const int M = BATCH * SEQ;                       // 4096
  float*  Q    = (float*)d_ws;                     // M*DIM f32
  float*  O    = Q + (size_t)M * DIM;              // M*DIM f32
  __bf16* Kbf  = (__bf16*)(O + (size_t)M * DIM);   // M*HD bf16
  __bf16* Vtbf = Kbf + (size_t)M * HD;             // BATCH*HD*SEQ bf16

  dim3 blk(256);

  gemm_bias_f32out<<<dim3(M / 128, DIM / 128), blk, 0, stream>>>(
      x, wq, bq, Q, M, DIM, DIM);
  gemm_bias_bf16out<<<dim3(M / 128, 1), blk, 0, stream>>>(
      x, wk, bk, Kbf, M, HD, DIM, 0);
  gemm_bias_bf16out<<<dim3(M / 128, 1), blk, 0, stream>>>(
      x, wv, bv, Vtbf, M, HD, DIM, 1);

  mqa_flash_attn<<<dim3(SEQ / 128, BATCH * NH), blk, 0, stream>>>(
      Q, Kbf, Vtbf, O);

  gemm_bias_f32out<<<dim3(M / 128, DIM / 128), blk, 0, stream>>>(
      O, wo, bo, out, M, DIM, DIM);
}
